// FreqSal_38414187495564
// MI455X (gfx1250) — matrix operand order repaired and checked by reading, not compiled, not run on hardware
//
#include <hip/hip_runtime.h>
#include <hip/hip_bf16.h>
#include <math.h>

#define PI_F 3.14159265358979323846f

typedef __attribute__((ext_vector_type(16))) _Float16 v16h;
typedef __attribute__((ext_vector_type(8)))  float    v8f;

#ifndef __has_builtin
#define __has_builtin(x) 0
#endif

#if defined(__gfx1250__) && __has_builtin(__builtin_amdgcn_global_load_async_to_lds_b32) && \
    __has_builtin(__builtin_amdgcn_s_wait_asynccnt)
#define USE_ASYNC_LDS 1
#else
#define USE_ASYNC_LDS 0
#endif

namespace fs {
constexpr int B = 16, DIM = 192, SIZE = 48, NF = 8;
constexpr int C2  = DIM / 2 + 1;   // 97
constexpr int HID = 48;
constexpr int FS  = SIZE / 2 + 1;  // 25
constexpr int HW  = SIZE * SIZE;   // 2304
constexpr int SP  = SIZE * FS;     // 1200
constexpr long S  = (long)B * DIM * HW;      // 7,077,888
constexpr long F2 = (long)B * DIM * SP * 2;  // 7,372,800
constexpr long FA = (long)B * DIM * SP;      // 3,686,400
constexpr long FC = (long)B * C2  * SP;      // 1,862,400
constexpr float EPS = 1e-5f;

// workspace layout (float offsets)
constexpr long o_rbn  = 0;
constexpr long o_tbn  = S;
constexpr long o_pool = 2 * S;
constexpr long o_route= 2 * S + 4096;
constexpr long o_ract = 2 * S + 32768;
constexpr long o_tact = o_ract + S;
constexpr long o_f1   = o_tact + S;     // row-FFT scratch; later: g
constexpr long o_a    = o_f1 + F2;      // amplitude; later: weights (spans a+p)
constexpr long o_p    = o_a + FA;
constexpr long o_ca   = o_p + FA;
constexpr long o_cp   = o_ca + FC;
constexpr long o_eh   = o_cp + FC;
constexpr long o_caE  = o_eh + FC;
constexpr long o_cpE  = o_caE + FC;
constexpr long o_pEh  = o_cpE + FC;
constexpr long o_pE   = o_pEh + FA;
constexpr long o_sc   = o_pE + FA;
constexpr long o_rts  = o_sc + FA;
constexpr long o_wgt  = o_a;            // reuse
constexpr long o_g    = o_f1;           // reuse
constexpr long o_out1 = o_ract;         // reuse
} // namespace fs

// ---------------------------------------------------------------------------
// BN (eval) for r and t + global-average pool of (r_bn + t_bn)
// ---------------------------------------------------------------------------
__global__ void bn_pool_kernel(const float* __restrict__ r, const float* __restrict__ t,
                               const float* __restrict__ bng, const float* __restrict__ bnb,
                               float* __restrict__ rbn, float* __restrict__ tbn,
                               float* __restrict__ pooled) {
  using namespace fs;
  __shared__ float red[256];
  const int bid = blockIdx.x;              // b*DIM + c
  const int c = bid % DIM;
  const float inv = rsqrtf(1.f + EPS);
  const float sr = bng[c] * inv,        br = bnb[c];
  const float st = bng[DIM + c] * inv,  bt = bnb[DIM + c];
  const long base = (long)bid * HW;
  float sum = 0.f;
  for (int i = threadIdx.x; i < HW; i += 256) {
    float rv = r[base + i] * sr + br;
    float tv = t[base + i] * st + bt;
    rbn[base + i] = rv;
    tbn[base + i] = tv;
    sum += rv + tv;
  }
  red[threadIdx.x] = sum;
  __syncthreads();
  for (int o = 128; o > 0; o >>= 1) {
    if (threadIdx.x < o) red[threadIdx.x] += red[threadIdx.x + o];
    __syncthreads();
  }
  if (threadIdx.x == 0) pooled[bid] = red[0] / (float)HW;
}

// ---------------------------------------------------------------------------
// Routing MLP (tiny; 1 block)
// ---------------------------------------------------------------------------
__global__ void route_kernel(const float* __restrict__ pooled, const float* __restrict__ fc1,
                             const float* __restrict__ fc2, const float* __restrict__ starS,
                             const float* __restrict__ starB, float* __restrict__ route) {
  using namespace fs;
  __shared__ float h[B * HID];
  const int t = threadIdx.x;
  const float s2 = starS[2], b2 = starB[2];
  for (int idx = t; idx < B * HID; idx += 256) {
    int b = idx / HID, o = idx % HID;
    float acc = 0.f;
    for (int k = 0; k < DIM; ++k) acc += pooled[b * DIM + k] * fc1[o * DIM + k];
    float rl = fmaxf(acc, 0.f);
    h[idx] = s2 * rl * rl + b2;
  }
  __syncthreads();
  for (int idx = t; idx < B * DIM; idx += 256) {
    int b = idx / DIM, d = idx % DIM;
    float l[NF];
    float mx = -1e30f;
    for (int f = 0; f < NF; ++f) {
      float acc = 0.f;
      const float* w = fc2 + (long)(f * DIM + d) * HID;
      for (int k = 0; k < HID; ++k) acc += h[b * HID + k] * w[k];
      l[f] = acc;
      mx = fmaxf(mx, acc);
    }
    float den = 0.f;
    for (int f = 0; f < NF; ++f) { l[f] = __expf(l[f] - mx); den += l[f]; }
    for (int f = 0; f < NF; ++f)
      route[((long)b * NF + f) * DIM + d] = l[f] / den;
  }
}

// ---------------------------------------------------------------------------
// Templated WMMA 1x1-conv GEMM: O[b,m,hw] = act( sum_k A[b,k,hw]*W[m,k] + bias )
// Compile-time dims (HWL,KD,MD) -> no integer division, fully unrolled K loop.
// Block: 128 thr (4 waves), tile = 64 rows x 32 out-channels, 2 WMMA/wave/Kstep.
// Staging: f32 tiles in LDS via async global->LDS loads (ASYNCcnt) when
// available, else synchronous loads (+prefetch). f16 fragments, f32 accum.
// act: 0=none, 1=LeakyReLU(0.1), 2=StarReLU(s*relu^2+b, scalars from device)
// ---------------------------------------------------------------------------
template <int HWL, int KD, int MD>
__global__ void gemm1x1_wmma(const float* __restrict__ A, const float* __restrict__ Wm,
                             const float* __restrict__ bias, float* __restrict__ O,
                             int act, const float* actS, const float* actB, int actIdx) {
  constexpr int NKB = (KD + 31) >> 5;
  __shared__ float ldsAf[64 * 32];   // rows x k
  __shared__ float ldsBf[32 * 32];   // k x n
  const int t = threadIdx.x;
  const int wave = t >> 5, lane = t & 31;
  const long row0 = (long)blockIdx.x * 64;
  const int m0 = blockIdx.y * 32;
  v8f acc0 = {}, acc1 = {};
#pragma unroll
  for (int kb = 0; kb < NKB; ++kb) {
    const int kbase = kb << 5;
    // ---- stage A tile (64 rows x 32 k), coalesced over hw ----
#pragma unroll
    for (int i = 0; i < 16; ++i) {
      int li = t + (i << 7);            // 0..2047
      int kk = li >> 6;                 // 0..31
      int rr = li & 63;                 // 0..63
      long grow = row0 + rr;
      int b  = (int)(grow / HWL);       // constexpr divisor -> mul/shift
      int hw = (int)(grow % HWL);
      int k = kbase + kk;
      int dst = rr * 32 + kk;
      if (k < KD) {
        const float* src = A + ((long)b * KD + k) * HWL + hw;
#if USE_ASYNC_LDS
        __builtin_amdgcn_global_load_async_to_lds_b32((int*)src, (int*)&ldsAf[dst], 0, 0);
#else
        ldsAf[dst] = *src;
#endif
      } else {
        ldsAf[dst] = 0.f;
      }
    }
    // ---- stage B tile (32 k x 32 n) ----
#pragma unroll
    for (int i = 0; i < 8; ++i) {
      int li = t + (i << 7);            // 0..1023
      int n  = li >> 5;                 // 0..31
      int kk = li & 31;
      int k = kbase + kk;
      int oc = m0 + n;
      int dst = kk * 32 + n;
      if (k < KD && oc < MD) {
        const float* src = Wm + (long)oc * KD + k;
#if USE_ASYNC_LDS
        __builtin_amdgcn_global_load_async_to_lds_b32((int*)src, (int*)&ldsBf[dst], 0, 0);
#else
        ldsBf[dst] = *src;
#endif
      } else {
        ldsBf[dst] = 0.f;
      }
    }
#if USE_ASYNC_LDS
    __builtin_amdgcn_s_wait_asynccnt(0);
#else
    if (kb + 1 < NKB) __builtin_prefetch(Wm + (long)m0 * KD + kbase + 32, 0, 1);
#endif
    __syncthreads();
    // ---- fragments (ISA 7.12.2: 16-bit A 16x32, B 32x16) ----
    union U16 { _Float16 e[16]; v16h v; } ua, ub0, ub1;
    {
      const int mrow = lane & 15, half = lane >> 4;
      const int rbase = (wave * 16 + mrow) * 32;
#pragma unroll
      for (int vi = 0; vi < 8; ++vi) {
        int k0 = ((vi >> 2) << 4) + (half << 3) + ((vi & 3) << 1);
        ua.e[2 * vi]     = (_Float16)ldsAf[rbase + k0];
        ua.e[2 * vi + 1] = (_Float16)ldsAf[rbase + k0 + 1];
      }
      const int kq = lane & 15;
#pragma unroll
      for (int vi = 0; vi < 8; ++vi) {
        int k0 = ((vi >> 2) << 4) + (half << 3) + ((vi & 3) << 1);
        ub0.e[2 * vi]     = (_Float16)ldsBf[k0 * 32 + kq];
        ub0.e[2 * vi + 1] = (_Float16)ldsBf[(k0 + 1) * 32 + kq];
        ub1.e[2 * vi]     = (_Float16)ldsBf[k0 * 32 + kq + 16];
        ub1.e[2 * vi + 1] = (_Float16)ldsBf[(k0 + 1) * 32 + kq + 16];
      }
    }
    acc0 = __builtin_amdgcn_wmma_f32_16x16x32_f16(false, ua.v, false, ub0.v,
                                                  (short)0, acc0, false, false);
    acc1 = __builtin_amdgcn_wmma_f32_16x16x32_f16(false, ua.v, false, ub1.v,
                                                  (short)0, acc1, false, false);
    __syncthreads();
  }
  // ---- epilogue: D layout -> col N = lane&15, row M = j + 8*(lane>>4) ----
  float s = 0.f, sb = 0.f;
  if (act == 2) { s = actS[actIdx]; sb = actB[actIdx]; }
  const int ncol = lane & 15, half = lane >> 4;
  union UA { v8f v; float e[8]; } uacc;
#pragma unroll
  for (int nh = 0; nh < 2; ++nh) {
    uacc.v = nh ? acc1 : acc0;
    const int oc = m0 + nh * 16 + ncol;
    if (oc < MD) {
      const float bv = bias ? bias[oc] : 0.f;
#pragma unroll
      for (int j = 0; j < 8; ++j) {
        long grow = row0 + wave * 16 + j + (half << 3);
        int b  = (int)(grow / HWL);
        int hw = (int)(grow % HWL);
        float v = uacc.e[j] + bv;
        if (act == 1)      v = (v >= 0.f) ? v : 0.1f * v;
        else if (act == 2) { float r_ = fmaxf(v, 0.f); v = s * r_ * r_ + sb; }
        O[((long)b * MD + oc) * HWL + hw] = v;
      }
    }
  }
}

// ---------------------------------------------------------------------------
// Spatial rfft along W (48 -> 25 complex), twiddle table in LDS
// ---------------------------------------------------------------------------
__global__ void rfft_w_kernel(const float* __restrict__ in, float* __restrict__ f1) {
  using namespace fs;
  __shared__ float ct[SIZE], st[SIZE];
  int t = threadIdx.x;
  if (t < SIZE) { float a = -2.f * PI_F * (float)t / (float)SIZE; ct[t] = __cosf(a); st[t] = __sinf(a); }
  __syncthreads();
  long idx = (long)blockIdx.x * 256 + t;
  if (idx >= FA) return;
  int wf = (int)(idx % FS);
  long row = idx / FS;                   // (b*DIM+d)*SIZE + h
  const float* src = in + row * SIZE;
  float re = 0.f, im = 0.f;
  for (int w = 0; w < SIZE; ++w) {
    float v = src[w];
    int k = (w * wf) % SIZE;
    re += v * ct[k];
    im += v * st[k];
  }
  f1[2 * idx] = re;
  f1[2 * idx + 1] = im;
}

// ---------------------------------------------------------------------------
// Complex DFT along H + ortho scale (1/48) + amplitude/phase epilogue
// ---------------------------------------------------------------------------
__global__ void fft_h_ap_kernel(const float* __restrict__ f1, float* __restrict__ amp,
                                float* __restrict__ pha) {
  using namespace fs;
  __shared__ float ct[SIZE], st[SIZE];
  int t = threadIdx.x;
  if (t < SIZE) { float a = -2.f * PI_F * (float)t / (float)SIZE; ct[t] = __cosf(a); st[t] = __sinf(a); }
  __syncthreads();
  long idx = (long)blockIdx.x * 256 + t;
  if (idx >= FA) return;
  int wf = (int)(idx % FS);
  long rem = idx / FS;
  int hf = (int)(rem % SIZE);
  long bd = rem / SIZE;
  float re = 0.f, im = 0.f;
  for (int h = 0; h < SIZE; ++h) {
    long j = (bd * SIZE + h) * FS + wf;
    float cr = f1[2 * j], ci = f1[2 * j + 1];
    int k = (h * hf) % SIZE;
    re += cr * ct[k] - ci * st[k];
    im += cr * st[k] + ci * ct[k];
  }
  const float sc = 1.f / 48.f;           // ortho 1/sqrt(48*48)
  re *= sc; im *= sc;
  amp[idx] = sqrtf(re * re + im * im);
  pha[idx] = atan2f(im, re);
}

// ---------------------------------------------------------------------------
// Channel rfft of amplitude (192 -> 97) + amplitude/phase epilogue
// ---------------------------------------------------------------------------
__global__ void cfft_kernel(const float* __restrict__ amp, float* __restrict__ ca,
                            float* __restrict__ cp) {
  using namespace fs;
  __shared__ float ct[DIM], st[DIM];
  int t = threadIdx.x;
  if (t < DIM) { float a = -2.f * PI_F * (float)t / (float)DIM; ct[t] = __cosf(a); st[t] = __sinf(a); }
  __syncthreads();
  long idx = (long)blockIdx.x * 256 + t;
  if (idx >= FC) return;
  int sp = (int)(idx % SP);
  long rem = idx / SP;
  int cf = (int)(rem % C2);
  int b  = (int)(rem / C2);
  float re = 0.f, im = 0.f;
  const float* src = amp + (long)b * DIM * SP + sp;
  for (int d = 0; d < DIM; ++d) {
    float v = src[(long)d * SP];
    int k = (d * cf) % DIM;
    re += v * ct[k];
    im += v * st[k];
  }
  const float sc = rsqrtf((float)DIM);
  re *= sc; im *= sc;
  ca[idx] = sqrtf(re * re + im * im);
  cp[idx] = atan2f(im, re);
}

// polar -> rect in place: (caE, cpE) -> (Re, Im)
__global__ void polar_kernel(float* __restrict__ caE, float* __restrict__ cpE) {
  using namespace fs;
  long idx = (long)blockIdx.x * 256 + threadIdx.x;
  if (idx >= FC) return;
  float a = caE[idx], p = cpE[idx];
  float sn, cs;
  __sincosf(p, &sn, &cs);
  caE[idx] = a * cs;
  cpE[idx] = a * sn;
}

// ---------------------------------------------------------------------------
// Channel irfft (97 -> 192, hermitian, ortho) fused with sc = c * amp
// ---------------------------------------------------------------------------
__global__ void cirfft_sc_kernel(const float* __restrict__ Xr, const float* __restrict__ Xi,
                                 const float* __restrict__ amp, float* __restrict__ sc) {
  using namespace fs;
  __shared__ float ct[DIM], st[DIM];
  int t = threadIdx.x;
  if (t < DIM) { float a = 2.f * PI_F * (float)t / (float)DIM; ct[t] = __cosf(a); st[t] = __sinf(a); }
  __syncthreads();
  long idx = (long)blockIdx.x * 256 + t;
  if (idx >= FA) return;
  int sp = (int)(idx % SP);
  long rem = idx / SP;
  int d = (int)(rem % DIM);
  int b = (int)(rem / DIM);
  const long base = (long)b * C2 * SP + sp;
  float acc = Xr[base];                         // cf = 0
  for (int cf = 1; cf < C2 - 1; ++cf) {
    float xr = Xr[base + (long)cf * SP], xi = Xi[base + (long)cf * SP];
    int k = (d * cf) % DIM;
    acc += 2.f * (xr * ct[k] - xi * st[k]);
  }
  float xr96 = Xr[base + (long)(C2 - 1) * SP];  // Nyquist
  acc += (d & 1) ? -xr96 : xr96;
  acc *= rsqrtf((float)DIM);
  sc[idx] = acc * amp[idx];
}

// rts (+)= sc * exp(i * pE)
__global__ void accum_rts_kernel(const float* __restrict__ sc, const float* __restrict__ pE,
                                 float* __restrict__ rts, int first) {
  using namespace fs;
  long idx = (long)blockIdx.x * 256 + threadIdx.x;
  if (idx >= FA) return;
  float s = sc[idx];
  float sn, cs;
  __sincosf(pE[idx], &sn, &cs);
  float re = s * cs, im = s * sn;
  if (first) { rts[2 * idx] = re;              rts[2 * idx + 1] = im; }
  else       { rts[2 * idx] += re;             rts[2 * idx + 1] += im; }
}

// weights[b,d,hf,wf] = sum_f route[b,f,d] * cw[hf,wf,f]
__global__ void weights_kernel(const float* __restrict__ route, const float* __restrict__ cw,
                               float* __restrict__ wgt) {
  using namespace fs;
  long idx = (long)blockIdx.x * 256 + threadIdx.x;
  if (idx >= FA) return;
  int sp = (int)(idx % SP);
  long rem = idx / SP;
  int d = (int)(rem % DIM);
  int b = (int)(rem / DIM);
  const float* cwp = cw + (long)sp * NF * 2;
  const float* rp  = route + (long)b * NF * DIM + d;
  float wr = 0.f, wi = 0.f;
#pragma unroll
  for (int f = 0; f < NF; ++f) {
    float rv = rp[(long)f * DIM];
    wr += rv * cwp[2 * f];
    wi += rv * cwp[2 * f + 1];
  }
  wgt[2 * idx] = wr;
  wgt[2 * idx + 1] = wi;
}

// g[b,d,h,wf] = sum_hf (rts * wgt)[b,d,hf,wf] * e^{+2pi i h hf / 48}
__global__ void ifft_h_kernel(const float* __restrict__ rts, const float* __restrict__ wgt,
                              float* __restrict__ g) {
  using namespace fs;
  __shared__ float ct[SIZE], st[SIZE];
  int t = threadIdx.x;
  if (t < SIZE) { float a = 2.f * PI_F * (float)t / (float)SIZE; ct[t] = __cosf(a); st[t] = __sinf(a); }
  __syncthreads();
  long idx = (long)blockIdx.x * 256 + t;
  if (idx >= FA) return;
  int wf = (int)(idx % FS);
  long rem = idx / FS;
  int h = (int)(rem % SIZE);
  long bd = rem / SIZE;
  float re = 0.f, im = 0.f;
  for (int hf = 0; hf < SIZE; ++hf) {
    long j = (bd * SIZE + hf) * FS + wf;
    float rr = rts[2 * j], ri = rts[2 * j + 1];
    float wr = wgt[2 * j], wi = wgt[2 * j + 1];
    float xr = rr * wr - ri * wi;
    float xi = rr * wi + ri * wr;
    int k = (h * hf) % SIZE;
    re += xr * ct[k] - xi * st[k];
    im += xr * st[k] + xi * ct[k];
  }
  g[2 * idx] = re;
  g[2 * idx + 1] = im;
}

// irfft along W (hermitian) + ortho scale + residuals -> out_1
__global__ void irfft_w_out1_kernel(const float* __restrict__ g, const float* __restrict__ rres,
                                    const float* __restrict__ tres, float* __restrict__ out1) {
  using namespace fs;
  __shared__ float ct[SIZE], st[SIZE];
  int t = threadIdx.x;
  if (t < SIZE) { float a = 2.f * PI_F * (float)t / (float)SIZE; ct[t] = __cosf(a); st[t] = __sinf(a); }
  __syncthreads();
  long idx = (long)blockIdx.x * 256 + t;
  if (idx >= S) return;
  int w = (int)(idx % SIZE);
  long row = idx / SIZE;                 // (b*DIM+d)*SIZE + h
  const long gbase = row * FS * 2;
  float acc = g[gbase];                  // k = 0
  for (int k = 1; k < FS - 1; ++k) {
    float gr = g[gbase + 2 * k], gi = g[gbase + 2 * k + 1];
    int a = (w * k) % SIZE;
    acc += 2.f * (gr * ct[a] - gi * st[a]);
  }
  float gN = g[gbase + 2 * (FS - 1)];    // Nyquist
  acc += (w & 1) ? -gN : gN;
  acc *= (1.f / 48.f);                   // ortho 1/sqrt(48*48)
  out1[idx] = acc + rres[idx] + tres[idx];
}

// BN + depthwise 3x3 (SAME, zero pad) + residual -> d_out
__global__ void dw_final_kernel(const float* __restrict__ out1, const float* __restrict__ bng,
                                const float* __restrict__ bnb, const float* __restrict__ dww,
                                float* __restrict__ out) {
  using namespace fs;
  long idx = (long)blockIdx.x * 256 + threadIdx.x;
  if (idx >= S) return;
  int w = (int)(idx % SIZE);
  long r1 = idx / SIZE;
  int h = (int)(r1 % SIZE);
  long bd = r1 / SIZE;
  int d = (int)(bd % DIM);
  const float scl = bng[d] * rsqrtf(1.f + EPS);
  const float bia = bnb[d];
  const long base = bd * HW;
  float acc = 0.f;
#pragma unroll
  for (int dh = -1; dh <= 1; ++dh) {
    int hh = h + dh;
    if (hh < 0 || hh >= SIZE) continue;
#pragma unroll
    for (int dw_ = -1; dw_ <= 1; ++dw_) {
      int ww = w + dw_;
      if (ww < 0 || ww >= SIZE) continue;
      float v = out1[base + hh * SIZE + ww] * scl + bia;
      acc += v * dww[d * 9 + (dh + 1) * 3 + (dw_ + 1)];
    }
  }
  out[idx] = acc + out1[idx];
}

// ---------------------------------------------------------------------------
template <int HWL, int KD, int MD>
static inline void launch_gemm(const float* A, const float* W, const float* bias, float* O,
                               int act, const float* aS, const float* aB, int ai,
                               hipStream_t stream) {
  dim3 grid((unsigned)(((long)fs::B * HWL) / 64), (unsigned)((MD + 31) / 32));
  gemm1x1_wmma<HWL, KD, MD><<<grid, dim3(128), 0, stream>>>(A, W, bias, O, act, aS, aB, ai);
}

extern "C" void kernel_launch(void* const* d_in, const int* in_sizes, int n_in,
                              void* d_out, int out_size, void* d_ws, size_t ws_size,
                              hipStream_t stream) {
  using namespace fs;
  (void)in_sizes; (void)n_in; (void)out_size; (void)ws_size;
  const float* r     = (const float*)d_in[0];
  const float* t     = (const float*)d_in[1];
  const float* bn_g  = (const float*)d_in[2];
  const float* bn_b  = (const float*)d_in[3];
  const float* crt_w = (const float*)d_in[4];
  const float* crt_b = (const float*)d_in[5];
  const float* starS = (const float*)d_in[6];
  const float* starB = (const float*)d_in[7];
  const float* fc1   = (const float*)d_in[8];
  const float* fc2   = (const float*)d_in[9];
  const float* cw    = (const float*)d_in[10];
  const float* ew1   = (const float*)d_in[11];
  const float* eb1   = (const float*)d_in[12];
  const float* ew2   = (const float*)d_in[13];
  const float* eb2   = (const float*)d_in[14];
  const float* pw1   = (const float*)d_in[15];
  const float* pb1   = (const float*)d_in[16];
  const float* pw2   = (const float*)d_in[17];
  const float* pb2   = (const float*)d_in[18];
  const float* dwg   = (const float*)d_in[19];
  const float* dwb   = (const float*)d_in[20];
  const float* dww   = (const float*)d_in[21];

  float* ws = (float*)d_ws;
  float* rbn   = ws + o_rbn;
  float* tbn   = ws + o_tbn;
  float* pool  = ws + o_pool;
  float* route = ws + o_route;
  float* ract  = ws + o_ract;
  float* tact  = ws + o_tact;
  float* f1    = ws + o_f1;
  float* a_    = ws + o_a;
  float* p_    = ws + o_p;
  float* ca_   = ws + o_ca;
  float* cp_   = ws + o_cp;
  float* eh    = ws + o_eh;
  float* caE   = ws + o_caE;
  float* cpE   = ws + o_cpE;
  float* pEh   = ws + o_pEh;
  float* pE_   = ws + o_pE;
  float* sc_   = ws + o_sc;
  float* rts   = ws + o_rts;
  float* wgt   = ws + o_wgt;
  float* g_    = ws + o_g;
  float* out1  = ws + o_out1;

  const int gA = (int)((FA + 255) / 256);   // 14400
  const int gC = (int)((FC + 255) / 256);   // 7275
  const int gS = (int)((S  + 255) / 256);   // 27648

  bn_pool_kernel<<<B * DIM, 256, 0, stream>>>(r, t, bn_g, bn_b, rbn, tbn, pool);
  route_kernel<<<1, 256, 0, stream>>>(pool, fc1, fc2, starS, starB, route);

  // conv_r / conv_t + StarReLU  (WMMA)
  launch_gemm<HW, DIM, DIM>(rbn, crt_w,             crt_b,       ract, 2, starS, starB, 0, stream);
  launch_gemm<HW, DIM, DIM>(tbn, crt_w + DIM * DIM, crt_b + DIM, tact, 2, starS, starB, 1, stream);

  auto branch = [&](const float* u, int ia, int ip, int iph, int first) {
    rfft_w_kernel<<<gA, 256, 0, stream>>>(u, f1);
    fft_h_ap_kernel<<<gA, 256, 0, stream>>>(f1, a_, p_);
    cfft_kernel<<<gC, 256, 0, stream>>>(a_, ca_, cp_);
    // enhance nets (WMMA, K=M=97 padded internally)
    launch_gemm<SP, C2, C2>(ca_, ew1 + (long)ia * C2 * C2, eb1 + ia * C2, eh,  1, starS, starB, 0, stream);
    launch_gemm<SP, C2, C2>(eh,  ew2 + (long)ia * C2 * C2, eb2 + ia * C2, caE, 0, starS, starB, 0, stream);
    launch_gemm<SP, C2, C2>(cp_, ew1 + (long)ip * C2 * C2, eb1 + ip * C2, eh,  1, starS, starB, 0, stream);
    launch_gemm<SP, C2, C2>(eh,  ew2 + (long)ip * C2 * C2, eb2 + ip * C2, cpE, 0, starS, starB, 0, stream);
    polar_kernel<<<gC, 256, 0, stream>>>(caE, cpE);
    cirfft_sc_kernel<<<gA, 256, 0, stream>>>(caE, cpE, a_, sc_);
    // phase net (WMMA)
    launch_gemm<SP, DIM, DIM>(p_,  pw1 + (long)iph * DIM * DIM, pb1 + iph * DIM, pEh, 1, starS, starB, 0, stream);
    launch_gemm<SP, DIM, DIM>(pEh, pw2 + (long)iph * DIM * DIM, pb2 + iph * DIM, pE_, 0, starS, starB, 0, stream);
    accum_rts_kernel<<<gA, 256, 0, stream>>>(sc_, pE_, rts, first);
  };
  branch(ract, 0, 1, 0, 1);
  branch(tact, 2, 3, 1, 0);

  weights_kernel<<<gA, 256, 0, stream>>>(route, cw, wgt);
  ifft_h_kernel<<<gA, 256, 0, stream>>>(rts, wgt, g_);
  irfft_w_out1_kernel<<<gS, 256, 0, stream>>>(g_, r, t, out1);
  dw_final_kernel<<<gS, 256, 0, stream>>>(out1, dwg, dwb, dww, (float*)d_out);
}
